// LinMHA_17111149707947
// MI455X (gfx1250) — matrix operand order repaired and checked by reading, not compiled, not run on hardware
//
#include <hip/hip_runtime.h>

#define DMODEL 512
#define HEADS 8
#define DKDIM 64
#define SEQ 8192
#define BATCH 4

typedef __attribute__((ext_vector_type(16))) __bf16 bfvec16;
typedef __attribute__((ext_vector_type(8)))  float  v8f;

union FragU { bfvec16 v; uint4 q[2]; };

__device__ __forceinline__ unsigned short f2bf(float f) {
  unsigned u = __float_as_uint(f);
  unsigned r = u + 0x7FFFu + ((u >> 16) & 1u);   // round-to-nearest-even
  return (unsigned short)(r >> 16);
}

// CDNA5 async global->LDS copy (16B per lane), tracked by ASYNCcnt.
__device__ __forceinline__ void async_copy_b128(unsigned lds_off, const void* gsrc) {
  asm volatile("global_load_async_to_lds_b128 %0, %1, off"
               :: "v"(lds_off), "v"(gsrc) : "memory");
}
__device__ __forceinline__ void wait_async0() {
  asm volatile("s_wait_asynccnt 0x0" ::: "memory");
}
__device__ __forceinline__ unsigned lds_off_of(const void* p) {
  return (unsigned)(size_t)p;   // LDS aperture: low 32 bits = wave-relative offset
}

// A-operand fragment (16xK rows): lane<16 -> row, K {kk..kk+7, kk+16..kk+23}
//                                 lane>=16 -> row, K {kk+8..kk+15, kk+24..kk+31}
__device__ __forceinline__ bfvec16 load_fragA(const unsigned short* lds, int row0,
                                              int kk, int ld, int lane) {
  const int r  = row0 + (lane & 15);
  const int kb = kk + ((lane >> 4) << 3);
  const unsigned short* p = lds + r * ld + kb;
  FragU f;
  f.q[0] = *(const uint4*)(p);
  f.q[1] = *(const uint4*)(p + 16);
  return f.v;
}

// B-operand fragment (Kx16, stored as rows of B^T): lane<16 -> col n, K {kk..kk+7, kk+16..kk+23}
//                                                   lane>=16 -> col n, K {kk+8..kk+15, kk+24..kk+31}
__device__ __forceinline__ bfvec16 load_fragB(const unsigned short* lds, int row0,
                                              int kk, int ld, int lane) {
  const int r  = row0 + (lane & 15);
  const int kb = kk + ((lane >> 4) << 3);
  const unsigned short* p = lds + r * ld + kb;
  FragU f;
  f.q[0] = *(const uint4*)(p);
  f.q[1] = *(const uint4*)(p + 16);
  return f.v;
}

#define WMMA_BF16(A, B, C) \
  __builtin_amdgcn_wmma_f32_16x16x32_bf16(false, (A), false, (B), (short)0, (C), false, false)

constexpr int LDA = 72;  // 64 + 8 pad (ushorts) to stagger LDS banks

__device__ __forceinline__ void cvt_store_tile(unsigned short* buf, int sr, int sc,
                                               const float4 (&f)[8]) {
#pragma unroll
  for (int j = 0; j < 8; ++j) {
    uint2 u;
    u.x = (unsigned)f2bf(f[j].x) | ((unsigned)f2bf(f[j].y) << 16);
    u.y = (unsigned)f2bf(f[j].z) | ((unsigned)f2bf(f[j].w) << 16);
    *(uint2*)&buf[sr * LDA + sc + j * 4] = u;
  }
}

__device__ __forceinline__ void mma_tile(const unsigned short* As, const unsigned short* Ws,
                                         int wm, int wn, int lane, v8f (&acc)[2][4]) {
#pragma unroll
  for (int kk = 0; kk < 64; kk += 32) {
    bfvec16 a0 = load_fragA(As, wm * 32 + 0,  kk, LDA, lane);
    bfvec16 a1 = load_fragA(As, wm * 32 + 16, kk, LDA, lane);
    bfvec16 w0 = load_fragB(Ws, wn * 64 + 0,  kk, LDA, lane);
    bfvec16 w1 = load_fragB(Ws, wn * 64 + 16, kk, LDA, lane);
    bfvec16 w2 = load_fragB(Ws, wn * 64 + 32, kk, LDA, lane);
    bfvec16 w3 = load_fragB(Ws, wn * 64 + 48, kk, LDA, lane);
    acc[0][0] = WMMA_BF16(a0, w0, acc[0][0]);
    acc[0][1] = WMMA_BF16(a0, w1, acc[0][1]);
    acc[0][2] = WMMA_BF16(a0, w2, acc[0][2]);
    acc[0][3] = WMMA_BF16(a0, w3, acc[0][3]);
    acc[1][0] = WMMA_BF16(a1, w0, acc[1][0]);
    acc[1][1] = WMMA_BF16(a1, w1, acc[1][1]);
    acc[1][2] = WMMA_BF16(a1, w2, acc[1][2]);
    acc[1][3] = WMMA_BF16(a1, w3, acc[1][3]);
  }
}

// ---------------------------------------------------------------------------
// C[M,N] = alpha * (A[M,K] @ W[N,K]^T) (+ bias).  Tile 128x128, K-step 64,
// double-buffered LDS. fp32 A converted through registers; bf16 A staged with
// async global->LDS copies overlapped with WMMA. W always fp32.
// ---------------------------------------------------------------------------
template <bool AIN_BF16, bool BF16OUT, bool BIAS>
__global__ __launch_bounds__(256) void gemm_xwT(const void* __restrict__ Ain,
                                                const float* __restrict__ W,
                                                const float* __restrict__ bias,
                                                void* __restrict__ Cout,
                                                int M, int N, int K, float alpha) {
  __shared__ __align__(16) unsigned short As[2][128 * LDA];
  __shared__ __align__(16) unsigned short Ws[2][128 * LDA];

  const int tid  = threadIdx.x;
  const int lane = tid & 31;
  const int wave = tid >> 5;        // 0..7
  const int wm   = wave >> 1;       // 0..3 : 32-row strip
  const int wn   = wave & 1;        // 0..1 : 64-col strip
  const int m0   = blockIdx.x * 128;
  const int n0   = blockIdx.y * 128;

  const int sr = tid >> 1;          // stage row 0..127
  const int sc = (tid & 1) * 32;    // stage col half

  const float* Wrow = W + (size_t)(n0 + sr) * K + sc;

  v8f acc[2][4] = {};
  float4 fa[8], fw[8];
  const int nk = K / 64;

  // ---- prologue: stage tile 0 ----
  if (AIN_BF16) {
    const unsigned short* ga = (const unsigned short*)Ain + (size_t)(m0 + sr) * K + sc;
#pragma unroll
    for (int j = 0; j < 4; ++j)
      async_copy_b128(lds_off_of(&As[0][sr * LDA + sc + j * 8]), ga + j * 8);
  } else {
    const float* ga = (const float*)Ain + (size_t)(m0 + sr) * K + sc;
#pragma unroll
    for (int j = 0; j < 8; ++j) fa[j] = *(const float4*)(ga + j * 4);
    cvt_store_tile(&As[0][0], sr, sc, fa);
  }
#pragma unroll
  for (int j = 0; j < 8; ++j) fw[j] = *(const float4*)(Wrow + j * 4);
  cvt_store_tile(&Ws[0][0], sr, sc, fw);

  for (int t = 0; t < nk; ++t) {
    const int cur = t & 1;
    if (AIN_BF16) wait_async0();          // our async writes to As[cur] landed
    __syncthreads();                      // everyone's tile t visible; tile t-1 reads done

    if (t + 1 < nk) {                     // start staging tile t+1 into buffer 1-cur
      const int k0 = (t + 1) * 64;
      if (AIN_BF16) {
        const unsigned short* ga =
            (const unsigned short*)Ain + (size_t)(m0 + sr) * K + k0 + sc;
#pragma unroll
        for (int j = 0; j < 4; ++j)
          async_copy_b128(lds_off_of(&As[1 - cur][sr * LDA + sc + j * 8]), ga + j * 8);
      } else {
        const float* ga = (const float*)Ain + (size_t)(m0 + sr) * K + k0 + sc;
#pragma unroll
        for (int j = 0; j < 8; ++j) fa[j] = *(const float4*)(ga + j * 4);
      }
#pragma unroll
      for (int j = 0; j < 8; ++j) fw[j] = *(const float4*)(Wrow + k0 + j * 4);
    }

    mma_tile(&As[cur][0], &Ws[cur][0], wm, wn, lane, acc);

    if (t + 1 < nk) {
      if (!AIN_BF16) cvt_store_tile(&As[1 - cur][0], sr, sc, fa);
      cvt_store_tile(&Ws[1 - cur][0], sr, sc, fw);
    }
  }

  // ---- epilogue: C layout lane<16 -> M=r, lane>=16 -> M=r+8; N = lane&15 ----
  const int colb = n0 + wn * 64 + (lane & 15);
  const int rhi  = (lane >> 4) * 8;
#pragma unroll
  for (int i = 0; i < 2; ++i) {
    const int rowb = m0 + wm * 32 + i * 16 + rhi;
#pragma unroll
    for (int j = 0; j < 4; ++j) {
      const int c = colb + j * 16;
      const float bv = BIAS ? bias[c] : 0.0f;
#pragma unroll
      for (int r = 0; r < 8; ++r) {
        const float val = acc[i][j][r] * alpha + bv;
        if (BF16OUT)
          ((unsigned short*)Cout)[(size_t)(rowb + r) * N + c] = f2bf(val);
        else
          ((float*)Cout)[(size_t)(rowb + r) * N + c] = val;
      }
    }
  }
}

// ---------------------------------------------------------------------------
// G[b,h] += Pk[b,:,h]^T @ Pv[b,:,h] over an S-chunk of 128 rows.
// Tiles transposed during LDS staging; partials via f32 global atomics.
// grid: (SEQ/128, BATCH*HEADS), block: 128 (4 waves).
// ---------------------------------------------------------------------------
__global__ __launch_bounds__(128) void g_accum(const unsigned short* __restrict__ Pk,
                                               const unsigned short* __restrict__ Pv,
                                               float* __restrict__ G) {
  constexpr int SC = 128, LDT = 136;  // 128 + 8 pad
  __shared__ __align__(16) unsigned short Kt[DKDIM * LDT];
  __shared__ __align__(16) unsigned short Vt[DKDIM * LDT];

  const int tid  = threadIdx.x;
  const int lane = tid & 31;
  const int wave = tid >> 5;          // 0..3
  const int bh   = blockIdx.y;        // 0..31
  const int b    = bh >> 3, h = bh & 7;
  const int s0   = blockIdx.x * SC;

  const size_t rowbase = ((size_t)b * SEQ + s0) * DMODEL + h * DKDIM;

  // ---- stage transposed: Kt[d][s], Vt[d][s] ----
  {
    const unsigned short* kp = Pk + rowbase + (size_t)tid * DMODEL;
    const unsigned short* vp = Pv + rowbase + (size_t)tid * DMODEL;
#pragma unroll
    for (int j = 0; j < 8; ++j) {
      union { uint4 q; unsigned short u[8]; } kq, vq;
      kq.q = *(const uint4*)(kp + j * 8);
      vq.q = *(const uint4*)(vp + j * 8);
#pragma unroll
      for (int e = 0; e < 8; ++e) {
        Kt[(j * 8 + e) * LDT + tid] = kq.u[e];
        Vt[(j * 8 + e) * LDT + tid] = vq.u[e];
      }
    }
  }
  __syncthreads();

  v8f acc[4] = {};
  const int drow = wave * 16;
#pragma unroll
  for (int kk = 0; kk < SC; kk += 32) {
    bfvec16 a = load_fragA(Kt, drow, kk, LDT, lane);
#pragma unroll
    for (int j = 0; j < 4; ++j) {
      bfvec16 bfr = load_fragB(Vt, j * 16, kk, LDT, lane);
      acc[j] = WMMA_BF16(a, bfr, acc[j]);
    }
  }

  float* g = G + (size_t)bh * DKDIM * DKDIM;
  const int e0  = lane & 15;
  const int rhi = (lane >> 4) * 8;
#pragma unroll
  for (int j = 0; j < 4; ++j)
#pragma unroll
    for (int r = 0; r < 8; ++r)
      atomicAdd(&g[(drow + r + rhi) * DKDIM + j * 16 + e0], acc[j][r]);
}

// ---------------------------------------------------------------------------
// Aout[rows, h] = Pq[rows, h] @ G[b,h]   (K = 64, bf16 out)
// Q slice staged via async global->LDS; G transposed fp32->bf16 through VALU.
// grid: (M/128, HEADS), block: 256 (8 waves).
// ---------------------------------------------------------------------------
__global__ __launch_bounds__(256) void apply_g(const unsigned short* __restrict__ Pq,
                                               const float* __restrict__ G,
                                               unsigned short* __restrict__ Aout) {
  constexpr int LDQ = 72;
  __shared__ __align__(16) unsigned short Qs[128 * LDQ];
  __shared__ __align__(16) unsigned short Gt[DKDIM * LDQ];

  const int tid  = threadIdx.x;
  const int lane = tid & 31;
  const int wave = tid >> 5;              // 0..7
  const int h    = blockIdx.y;
  const int row0 = blockIdx.x * 128;      // global row in [0, B*S)
  const int b    = row0 / SEQ;

  // ---- stage Q slice [128 x 64] via async copies ----
  {
    const int r = tid >> 1, c0 = (tid & 1) * 32;
    const unsigned short* qp = Pq + (size_t)(row0 + r) * DMODEL + h * DKDIM + c0;
#pragma unroll
    for (int j = 0; j < 4; ++j)
      async_copy_b128(lds_off_of(&Qs[r * LDQ + c0 + j * 8]), qp + j * 8);
  }
  // ---- stage G transposed: Gt[e][d] = bf16(G[d][e]) ----
  {
    const float* g  = G + (size_t)(b * HEADS + h) * DKDIM * DKDIM;
    const int d  = tid >> 2;
    const int e0 = (tid & 3) * 16;
#pragma unroll
    for (int j = 0; j < 4; ++j) {
      float4 f = *(const float4*)(g + d * DKDIM + e0 + j * 4);
      Gt[(e0 + j * 4 + 0) * LDQ + d] = f2bf(f.x);
      Gt[(e0 + j * 4 + 1) * LDQ + d] = f2bf(f.y);
      Gt[(e0 + j * 4 + 2) * LDQ + d] = f2bf(f.z);
      Gt[(e0 + j * 4 + 3) * LDQ + d] = f2bf(f.w);
    }
  }
  wait_async0();
  __syncthreads();

  v8f acc[4] = {};
  const int m = wave * 16;
#pragma unroll
  for (int kk = 0; kk < 64; kk += 32) {
    bfvec16 a = load_fragA(Qs, m, kk, LDQ, lane);
#pragma unroll
    for (int j = 0; j < 4; ++j) {
      bfvec16 bfr = load_fragB(Gt, j * 16, kk, LDQ, lane);
      acc[j] = WMMA_BF16(a, bfr, acc[j]);
    }
  }

  const int e0s = lane & 15;
  const int rhi = (lane >> 4) * 8;
#pragma unroll
  for (int j = 0; j < 4; ++j)
#pragma unroll
    for (int r = 0; r < 8; ++r)
      Aout[(size_t)(row0 + m + r + rhi) * DMODEL + h * DKDIM + j * 16 + e0s] =
          f2bf(acc[j][r]);
}

__global__ void zero_f(float* p, int n) {
  int i = blockIdx.x * blockDim.x + threadIdx.x;
  if (i < n) p[i] = 0.0f;
}

// ---------------------------------------------------------------------------
extern "C" void kernel_launch(void* const* d_in, const int* in_sizes, int n_in,
                              void* d_out, int out_size, void* d_ws, size_t ws_size,
                              hipStream_t stream) {
  (void)in_sizes; (void)n_in; (void)out_size; (void)ws_size;

  const float* q  = (const float*)d_in[0];
  const float* k  = (const float*)d_in[1];
  const float* v  = (const float*)d_in[2];
  const float* Wq = (const float*)d_in[3];
  const float* Wk = (const float*)d_in[4];
  const float* Wv = (const float*)d_in[5];
  const float* Wo = (const float*)d_in[6];
  const float* bo = (const float*)d_in[7];

  const int M = BATCH * SEQ;  // 32768
  char* ws = (char*)d_ws;
  const size_t projBytes = (size_t)M * DMODEL * sizeof(unsigned short);
  unsigned short* Pq = (unsigned short*)ws; ws += projBytes;
  unsigned short* Pk = (unsigned short*)ws; ws += projBytes;
  unsigned short* Pv = (unsigned short*)ws; ws += projBytes;
  unsigned short* Ao = (unsigned short*)ws; ws += projBytes;
  float* G = (float*)ws;
  const int gElems = BATCH * HEADS * DKDIM * DKDIM;  // 131072

  const float scale = 0.125f;  // dk^-0.5, folded into Pq and Pk
  const dim3 gGemm(M / 128, DMODEL / 128);

  zero_f<<<(gElems + 255) / 256, 256, 0, stream>>>(G, gElems);

  gemm_xwT<false, true, false><<<gGemm, 256, 0, stream>>>(q, Wq, nullptr, Pq,
                                                          M, DMODEL, DMODEL, scale);
  gemm_xwT<false, true, false><<<gGemm, 256, 0, stream>>>(k, Wk, nullptr, Pk,
                                                          M, DMODEL, DMODEL, scale);
  gemm_xwT<false, true, false><<<gGemm, 256, 0, stream>>>(v, Wv, nullptr, Pv,
                                                          M, DMODEL, DMODEL, 1.0f);

  g_accum<<<dim3(SEQ / 128, BATCH * HEADS), 128, 0, stream>>>(Pk, Pv, G);

  apply_g<<<dim3(M / 128, HEADS), 256, 0, stream>>>(Pq, G, Ao);

  gemm_xwT<true, false, true><<<gGemm, 256, 0, stream>>>(Ao, Wo, bo, (float*)d_out,
                                                         M, DMODEL, DMODEL, 1.0f);
}
